// Param_Dist_66468913872924
// MI455X (gfx1250) — matrix operand order repaired and checked by reading, not compiled, not run on hardware
//
#include <hip/hip_runtime.h>
#include <hip/hip_bf16.h>
#include <stdint.h>

// ---------------------------------------------------------------------------
// out[i,j] = yw[j] - (x @ W^T)[i,j],  yw[j] = sum_k y[j,k]*W[j,k]
// N = M = D = 4096, fp32 in / fp32 out.
//   1) fp32 -> bf16 convert of x, W into d_ws (streaming)
//   2) yw reduction (streaming)
//   3) bf16 WMMA GEMM, fp32 accumulate. Block tile 128x256, K-panel 64,
//      TDM (tensor_load_to_lds) double-buffered LDS staging.
//      8 waves as 2x4; each wave computes 64x64 = 4x4 WMMA tiles
//      -> 1.0 ds_load_b128 per v_wmma (A and B each reused 4x).
// ---------------------------------------------------------------------------

#define DIM 4096
#define BM 128
#define BN 256
#define BK 64
#define KSTEPS (DIM / BK)   // 64

typedef __bf16 bf16x8  __attribute__((ext_vector_type(8)));
typedef __bf16 bf16x16 __attribute__((ext_vector_type(16)));
typedef float  f32x8   __attribute__((ext_vector_type(8)));
typedef unsigned int u32x4 __attribute__((ext_vector_type(4)));
typedef int    i32x4   __attribute__((ext_vector_type(4)));
typedef int    i32x8   __attribute__((ext_vector_type(8)));

// ---------------- fp32 -> bf16 (RNE) conversion kernel ---------------------

__device__ __forceinline__ unsigned f32_to_bf16_bits(float f) {
    unsigned u = __float_as_uint(f);
    unsigned r = u + 0x7FFFu + ((u >> 16) & 1u);
    if ((u & 0x7F800000u) == 0x7F800000u) r = u;   // keep inf/nan payload
    return r >> 16;
}

__global__ void __launch_bounds__(256)
convert_bf16_kernel(const float* __restrict__ src, unsigned short* __restrict__ dst, int n4) {
    int i = blockIdx.x * blockDim.x + threadIdx.x;
    if (i >= n4) return;
    float4 f = reinterpret_cast<const float4*>(src)[i];
    uint2 p;
    p.x = f32_to_bf16_bits(f.x) | (f32_to_bf16_bits(f.y) << 16);
    p.y = f32_to_bf16_bits(f.z) | (f32_to_bf16_bits(f.w) << 16);
    reinterpret_cast<uint2*>(dst)[i] = p;
}

// ---------------- yw[j] = sum_k y[j,k] * W[j,k] ----------------------------

__global__ void __launch_bounds__(256)
yw_kernel(const float* __restrict__ y, const float* __restrict__ W, float* __restrict__ yw) {
    __shared__ float red[256];
    const int j   = blockIdx.x;
    const int tid = threadIdx.x;
    const float4* yr = reinterpret_cast<const float4*>(y + (size_t)j * DIM);
    const float4* wr = reinterpret_cast<const float4*>(W + (size_t)j * DIM);
    float acc = 0.0f;
    for (int k = tid; k < DIM / 4; k += 256) {
        float4 a = yr[k];
        float4 b = wr[k];
        acc += a.x * b.x + a.y * b.y + a.z * b.z + a.w * b.w;
    }
    red[tid] = acc;
    __syncthreads();
    for (int s = 128; s > 0; s >>= 1) {
        if (tid < s) red[tid] += red[tid + s];
        __syncthreads();
    }
    if (tid == 0) yw[j] = red[0];
}

// ---------------- TDM: 2D tile (rows x kelems of bf16) -> LDS --------------
// D# per CDNA5 ISA. 2-byte elements, row-major tensor
// with row stride `stride_elems`; tile written densely into LDS.

__device__ __forceinline__ void tdm_load_tile_bf16(
    unsigned lds_byte_addr, const unsigned short* gptr,
    unsigned tile_rows, unsigned tile_kelems,
    unsigned tensor_d0, unsigned tensor_d1, unsigned long long stride_elems)
{
    unsigned long long ga = (unsigned long long)(uintptr_t)gptr;
    u32x4 g0;
    g0.x = 1u;                                               // count=1 (valid), user mode
    g0.y = lds_byte_addr;                                    // lds_addr
    g0.z = (unsigned)(ga & 0xFFFFFFFFu);                     // global_addr[31:0]
    g0.w = (unsigned)((ga >> 32) & 0x1FFFFFFu) | (2u << 30); // addr[56:32] | type=2

    i32x8 g1;
    g1[0] = (int)(1u << 16);                                 // data_size=1 (2B), no multicast
    g1[1] = (int)((tensor_d0 & 0xFFFFu) << 16);              // tensor_dim0[15:0]
    g1[2] = (int)(((tensor_d0 >> 16) & 0xFFFFu) |
                  ((tensor_d1 & 0xFFFFu) << 16));            // dim0 hi | dim1 lo
    g1[3] = (int)(((tensor_d1 >> 16) & 0xFFFFu) |
                  ((tile_kelems & 0xFFFFu) << 16));          // dim1 hi | tile_dim0
    g1[4] = (int)(tile_rows & 0xFFFFu);                      // tile_dim1 (tile_dim2=0)
    g1[5] = (int)(unsigned)(stride_elems & 0xFFFFFFFFu);     // dim0_stride[31:0]
    g1[6] = (int)(unsigned)((stride_elems >> 32) & 0xFFFFu); // dim0_stride[47:32]
    g1[7] = 0;

    i32x4 g2;
    g2[0] = 1;  // tensor_dim2 = 1
    g2[1] = 1;  // tensor_dim3 = 1 (iterate disabled)
    g2[2] = 0;
    g2[3] = 0;  // tile_dim3 = 0 (unused)
    i32x4 g3 = {0, 0, 0, 0};

#if defined(__clang_major__) && (__clang_major__ >= 23)
    i32x8 g4 = {0, 0, 0, 0, 0, 0, 0, 0};
    __builtin_amdgcn_tensor_load_to_lds(g0, g1, g2, g3, g4, 0);
#else
    __builtin_amdgcn_tensor_load_to_lds(g0, g1, g2, g3, 0);
#endif
}

// ---------------- bf16 WMMA GEMM with fused epilogue -----------------------

__global__ void __launch_bounds__(256)
param_dist_gemm_kernel(const unsigned short* __restrict__ Abf,   // x in bf16, (N,D)
                       const unsigned short* __restrict__ Bbf,   // W in bf16, (M,D)
                       const float* __restrict__ yw,
                       float* __restrict__ out)
{
    __shared__ __align__(16) unsigned short sA[2][BM * BK];   // 2 x 16 KB
    __shared__ __align__(16) unsigned short sB[2][BN * BK];   // 2 x 32 KB

    const int wave = threadIdx.x >> 5;
    const int lane = threadIdx.x & 31;
    const int wm   = wave >> 2;          // 0..1  (row band of 64)
    const int wn   = wave & 3;           // 0..3  (col band of 64)
    const int hi   = lane >> 4;          // half-wave select
    const int r    = lane & 15;          // row / col within 16

    const int rowBase = blockIdx.y * BM;
    const int colBase = blockIdx.x * BN;

    f32x8 acc[4][4];
#pragma unroll
    for (int i = 0; i < 4; ++i)
#pragma unroll
        for (int j = 0; j < 4; ++j) acc[i][j] = (f32x8)0.0f;

    const unsigned ldsA0 = (unsigned)(uintptr_t)&sA[0][0];
    const unsigned ldsA1 = (unsigned)(uintptr_t)&sA[1][0];
    const unsigned ldsB0 = (unsigned)(uintptr_t)&sB[0][0];
    const unsigned ldsB1 = (unsigned)(uintptr_t)&sB[1][0];

    // Preload K-panel 0 into buffer 0 (wave 0 drives the TDM).
    if (wave == 0) {
        tdm_load_tile_bf16(ldsA0, Abf + (size_t)rowBase * DIM, BM, BK, DIM, DIM, DIM);
        tdm_load_tile_bf16(ldsB0, Bbf + (size_t)colBase * DIM, BN, BK, DIM, DIM, DIM);
        __builtin_amdgcn_s_wait_tensorcnt(0);
    }
    __syncthreads();

    for (int kk = 0; kk < KSTEPS; ++kk) {
        const int cur = kk & 1;

        // Issue next panel's TDM into the other buffer while we compute.
        if (wave == 0 && (kk + 1) < KSTEPS) {
            const unsigned la = cur ? ldsA0 : ldsA1;
            const unsigned lb = cur ? ldsB0 : ldsB1;
            const size_t koff = (size_t)(kk + 1) * BK;
            tdm_load_tile_bf16(la, Abf + (size_t)rowBase * DIM + koff, BM, BK, DIM, DIM, DIM);
            tdm_load_tile_bf16(lb, Bbf + (size_t)colBase * DIM + koff, BN, BK, DIM, DIM, DIM);
        }

        // Two WMMA sub-steps of K=32 over the 64-wide LDS panel.
#pragma unroll
        for (int s = 0; s < 2; ++s) {
            const int ks = s * 32;   // ushort offset within a row

            // A fragments (16x32 bf16): lane holds row r; VGPR0-3 = K 8*hi..+7,
            // VGPR4-7 = K 16+8*hi..+7  -> two ds_load_b128 per fragment.
            bf16x16 afrag[4];
#pragma unroll
            for (int t = 0; t < 4; ++t) {
                const int rowOff = (wm * 64 + t * 16 + r) * BK + ks;   // ushort units
                bf16x8 lo = *reinterpret_cast<const bf16x8*>(&sA[cur][rowOff + hi * 8]);
                bf16x8 hh = *reinterpret_cast<const bf16x8*>(&sA[cur][rowOff + 16 + hi * 8]);
                afrag[t] = __builtin_shufflevector(lo, hh, 0,1,2,3,4,5,6,7,8,9,10,11,12,13,14,15);
            }

            // B fragments (32x16 bf16): lane holds col r; K = 8*hi..+7, then 16+8*hi..+7.
            bf16x16 bfrag[4];
#pragma unroll
            for (int t = 0; t < 4; ++t) {
                const int rowOff = (wn * 64 + t * 16 + r) * BK + ks;   // ushort units
                bf16x8 lo = *reinterpret_cast<const bf16x8*>(&sB[cur][rowOff + hi * 8]);
                bf16x8 hh = *reinterpret_cast<const bf16x8*>(&sB[cur][rowOff + 16 + hi * 8]);
                bfrag[t] = __builtin_shufflevector(lo, hh, 0,1,2,3,4,5,6,7,8,9,10,11,12,13,14,15);
            }

#pragma unroll
            for (int i = 0; i < 4; ++i)
#pragma unroll
                for (int j = 0; j < 4; ++j)
                    acc[i][j] = __builtin_amdgcn_wmma_f32_16x16x32_bf16(
                        false, afrag[i], false, bfrag[j],
                        (short)0, acc[i][j], false, false);
        }

        if (wave == 0 && (kk + 1) < KSTEPS)
            __builtin_amdgcn_s_wait_tensorcnt(0);   // next panel landed
        __syncthreads();                            // publish LDS / retire reads
    }

    // Epilogue: C/D layout — VGPR v holds row (v + 8*hi), col = lane&15.
#pragma unroll
    for (int i = 0; i < 4; ++i) {
#pragma unroll
        for (int j = 0; j < 4; ++j) {
            const int col = colBase + wn * 64 + j * 16 + r;
            const float ywv = yw[col];
            float* o = out + (size_t)(rowBase + wm * 64 + i * 16 + 8 * hi) * DIM + col;
#pragma unroll
            for (int v = 0; v < 8; ++v)
                o[(size_t)v * DIM] = ywv - acc[i][j][v];
        }
    }
}

// ---------------------------------------------------------------------------

extern "C" void kernel_launch(void* const* d_in, const int* in_sizes, int n_in,
                              void* d_out, int out_size, void* d_ws, size_t ws_size,
                              hipStream_t stream) {
    const float* x = (const float*)d_in[0];   // (N, D)
    const float* y = (const float*)d_in[1];   // (M, D)
    const float* W = (const float*)d_in[2];   // (M, D) as used by einsum('ik,jk->ij')
    float* out = (float*)d_out;

    const size_t elems = (size_t)DIM * DIM;           // 16M
    unsigned short* xbf = (unsigned short*)d_ws;                        // 32 MB
    unsigned short* wbf = (unsigned short*)((char*)d_ws + elems * 2);   // 32 MB
    float*          yw  = (float*)((char*)d_ws + elems * 4);            // 16 KB

    // 1) fp32 -> bf16 for both GEMM operands
    const int n4 = (int)(elems / 4);
    convert_bf16_kernel<<<dim3((n4 + 255) / 256), dim3(256), 0, stream>>>(x, xbf, n4);
    convert_bf16_kernel<<<dim3((n4 + 255) / 256), dim3(256), 0, stream>>>(W, wbf, n4);

    // 2) yw[j] = sum_k y[j,k] * W[j,k]
    yw_kernel<<<dim3(DIM), dim3(256), 0, stream>>>(y, W, yw);

    // 3) out = yw[j] - x @ W^T  (bf16 WMMA, fp32 accumulate)
    dim3 grid(DIM / BN, DIM / BM);
    param_dist_gemm_kernel<<<grid, dim3(256), 0, stream>>>(xbf, wbf, yw, out);
}
